// GraphSPNMargkAry_69956427317253
// MI455X (gfx1250) — matrix operand order repaired and checked by reading, not compiled, not run on hardware
//
#include <hip/hip_runtime.h>
#include <hip/hip_bf16.h>
#include <math.h>

// ---------------- problem constants (from reference) ----------------
#define ND_N      9
#define ARITY     5
#define NK        5
#define NI        40          // mixture components
#define NDIM      30          // 5 rows of [x, a*5] interleaved
#define NCOMBO    126         // C(9,5)
#define EMPTY_TOK 4
#define KTOT      150         // NDIM * NK
#define ROWS_PB   128         // 126 combos padded to 8 wave-tiles of 16
#define BSW_ELEMS (5*3*32*16) // pre-swizzled B halves (15360 B)

typedef __attribute__((ext_vector_type(16))) _Float16 v16h;
typedef __attribute__((ext_vector_type(8)))  _Float16 v8h;
typedef __attribute__((ext_vector_type(8)))  float    v8f;

// workspace byte offsets
#define WS_COMBOS 0           // int[126] (packed 4-bit x5)
#define WS_LOGW   4096        // float[40]
#define WS_BSW    8192        // _Float16[BSW_ELEMS]

// -------------------------------------------------------------------
// place one-hot(tok) (5 bits) at global bit 5*d of the 150-bit mask.
// d is a compile-time constant after unrolling -> word/shift fold.
__device__ __forceinline__ void placebits(unsigned* mw, int d, int tok) {
  const unsigned oh = ((unsigned)tok < (unsigned)NK) ? (1u << tok) : 0u;
  const int bit = 5 * d;
  const int w   = bit >> 5;
  const int sh  = bit & 31;
  mw[w] |= oh << sh;
  if (sh > 27 && w < 4) mw[w + 1] |= oh >> (32 - sh);
}

// -------------------------------------------------------------------
// Stage 1: packed combos, log_softmax(w), and B = log_softmax(theta)^T
// pre-swizzled into the CDNA5 16-bit B-matrix fragment order:
//   element ((s*3+tile)*32 + lane)*16 + h  holds
//   B[K = (h&7) + 8*(lane/16) + 16*(h/8) + 32*s][n = tile*16 + lane%16]
__global__ void spn_prep_kernel(const float* __restrict__ theta,   // [40][30][5]
                                const float* __restrict__ wlog,    // [40]
                                int*   __restrict__ combos,        // [126] packed
                                float* __restrict__ lw,            // [40]
                                _Float16* __restrict__ bsw) {
  const int t = threadIdx.x;

  if (t == 0) {  // lexicographic, matches itertools.combinations(range(9), 5)
    int idx = 0;
    for (int c0 = 0; c0 < ND_N; ++c0)
      for (int c1 = c0 + 1; c1 < ND_N; ++c1)
        for (int c2 = c1 + 1; c2 < ND_N; ++c2)
          for (int c3 = c2 + 1; c3 < ND_N; ++c3)
            for (int c4 = c3 + 1; c4 < ND_N; ++c4)
              combos[idx++] = c0 | (c1 << 4) | (c2 << 8) | (c3 << 12) | (c4 << 16);
  }

  if (t < NI) {
    float m = -INFINITY;
    for (int i = 0; i < NI; ++i) m = fmaxf(m, wlog[i]);
    float s = 0.f;
    for (int i = 0; i < NI; ++i) s += __expf(wlog[i] - m);
    lw[t] = wlog[t] - (m + __logf(s));
  }

  for (int e = t; e < BSW_ELEMS; e += blockDim.x) {
    const int h    = e & 15;
    const int lane = (e >> 4) & 31;
    const int rest = e >> 9;          // s*3 + tile
    const int tile = rest % 3;
    const int s    = rest / 3;
    const int K    = (h & 7) + 8 * (lane >> 4) + 16 * (h >> 3) + 32 * s;
    const int n    = tile * 16 + (lane & 15);
    float val = 0.f;
    if (K < KTOT && n < NI) {
      const int d   = K / NK;
      const int tok = K % NK;
      const float* row = theta + (n * NDIM + d) * NK;
      float m = row[0];
      for (int k = 1; k < NK; ++k) m = fmaxf(m, row[k]);
      float ss = 0.f;
      for (int k = 0; k < NK; ++k) ss += __expf(row[k] - m);
      val = row[tok] - (m + __logf(ss));
    }
    bsw[e] = (_Float16)val;
  }
}

// -------------------------------------------------------------------
// one K-step: A fragment via LDS bit->half expansion table (2 ds_load_b128),
// B fragments from LDS, 3 WMMAs.
template <int S>
__device__ __forceinline__ void do_kstep(const unsigned* mw, int group, int lane,
                                         const _Float16* BsLDS, const _Float16* TsLDS,
                                         v8f& acc0, v8f& acc1, v8f& acc2) {
  // A layout: lane half h holds A[M=lane%16][K=(h<8?h:h+8)+8*group+32*S]
  // -> bits {0..7} and {16..23} of (mw[S] >> 8*group).
  const unsigned mg = mw[S] >> (group * 8);
  const v8h lo = *(const v8h*)(TsLDS + (mg & 0xFFu) * 8);
  const v8h hi = *(const v8h*)(TsLDS + ((mg >> 16) & 0xFFu) * 8);
  const v16h af = __builtin_shufflevector(lo, hi, 0, 1, 2, 3, 4, 5, 6, 7,
                                          8, 9, 10, 11, 12, 13, 14, 15);
  const _Float16* bb = BsLDS + (((S * 3) * 32) + lane) * 16;
  const v16h b0 = *(const v16h*)(bb);
  const v16h b1 = *(const v16h*)(bb + 32 * 16);
  const v16h b2 = *(const v16h*)(bb + 64 * 16);
  acc0 = __builtin_amdgcn_wmma_f32_16x16x32_f16(false, af, false, b0, (short)0, acc0, false, false);
  acc1 = __builtin_amdgcn_wmma_f32_16x16x32_f16(false, af, false, b1, (short)0, acc1, false, false);
  acc2 = __builtin_amdgcn_wmma_f32_16x16x32_f16(false, af, false, b2, (short)0, acc2, false, false);
}

// -------------------------------------------------------------------
// Stage 2: one block per batch graph; 8 waves x 16-combo tiles x 48 cols.
__global__ __launch_bounds__(256)
void spn_main_kernel(const int* __restrict__ xg,     // [B][9]
                     const int* __restrict__ ag,     // [B][9][9]
                     const float* __restrict__ rate, // [1]
                     const int* __restrict__ combos, // [126] packed
                     const float* __restrict__ lw,   // [40]
                     const _Float16* __restrict__ bsw,
                     float* __restrict__ out) {      // [B]
  __shared__ __align__(32) _Float16 Bs[BSW_ELEMS];   // 15360 B
  __shared__ __align__(16) _Float16 Ts[256 * 8];     // 4 KB bit->half table
  __shared__ int   xs[ND_N];
  __shared__ int   as[ND_N * ND_N];
  __shared__ float lsh[ROWS_PB];

  const int b   = blockIdx.x;
  const int tid = threadIdx.x;

  // async global->LDS copy of the 15 KB B table (960 x 16B chunks)
  {
    const uint32_t lds_base = (uint32_t)(uintptr_t)&Bs[0];
    const uint64_t gbase    = (uint64_t)(uintptr_t)bsw;
    for (int t = tid; t < BSW_ELEMS / 8; t += 256) {
      const uint32_t dst = lds_base + (uint32_t)t * 16u;
      const uint32_t src = (uint32_t)t * 16u;
      asm volatile("global_load_async_to_lds_b128 %0, %1, %2 offset:0"
                   :: "v"(dst), "v"(src), "s"(gbase) : "memory");
    }
  }
  // bit->half expansion table: Ts[v][j] = (v>>j)&1 as f16
#pragma unroll
  for (int j = 0; j < 8; ++j)
    Ts[tid * 8 + j] = ((tid >> j) & 1) ? (_Float16)1.0f : (_Float16)0.0f;

  for (int t = tid; t < ND_N * ND_N; t += 256) as[t] = ag[b * 81 + t];
  if (tid < ND_N) xs[tid] = xg[b * ND_N + tid];
  asm volatile("s_wait_asynccnt 0" ::: "memory");
  __syncthreads();

  const int wave  = tid >> 5;
  const int lane  = tid & 31;
  const int col   = lane & 15;
  const int group = lane >> 4;

  // 150-bit one-hot row mask for this lane's combo (A-matrix row = col)
  const int nc = wave * 16 + col;
  unsigned mw[5] = {0u, 0u, 0u, 0u, 0u};
  if (nc < NCOMBO) {
    const int cw = combos[nc];
    int c[ARITY];
#pragma unroll
    for (int j = 0; j < ARITY; ++j) c[j] = (cw >> (4 * j)) & 0xF;
#pragma unroll
    for (int j = 0; j < ARITY; ++j) {
      placebits(mw, j * 6, xs[c[j]]);                         // d = j*6 : x token
#pragma unroll
      for (int k = 0; k < ARITY; ++k)
        placebits(mw, j * 6 + 1 + k, as[c[j] * ND_N + c[k]]); // d = j*6+1+k : a token
    }
  } // else: mask stays 0 -> zero A row -> comp = 0 (excluded later)

  v8f acc0 = {}, acc1 = {}, acc2 = {};
  do_kstep<0>(mw, group, lane, Bs, Ts, acc0, acc1, acc2);
  do_kstep<1>(mw, group, lane, Bs, Ts, acc0, acc1, acc2);
  do_kstep<2>(mw, group, lane, Bs, Ts, acc0, acc1, acc2);
  do_kstep<3>(mw, group, lane, Bs, Ts, acc0, acc1, acc2);
  do_kstep<4>(mw, group, lane, Bs, Ts, acc0, acc1, acc2);

  // C/D layout: lane holds comp[M = r + 8*group][i = tile*16 + col] in acc{tile}[r]
  const float lw0 = lw[col];
  const float lw1 = lw[16 + col];
  const float lw2 = (col < 8) ? lw[32 + col] : 0.f;

#pragma unroll
  for (int r = 0; r < 8; ++r) {
    const float a0 = acc0[r] + lw0;
    const float a1 = acc1[r] + lw1;
    const float a2 = (col < 8) ? (acc2[r] + lw2) : -INFINITY; // i = 32+col < 40
    // branchless two-pass logsumexp across the 16-lane group
    float pm = fmaxf(fmaxf(a0, a1), a2);
#pragma unroll
    for (int off = 1; off <= 8; off <<= 1)
      pm = fmaxf(pm, __shfl_xor(pm, off, 32));
    float ps = __expf(a0 - pm) + __expf(a1 - pm) + __expf(a2 - pm); // exp(-inf)=0
#pragma unroll
    for (int off = 1; off <= 8; off <<= 1)
      ps += __shfl_xor(ps, off, 32);
    if (col == 0) {
      const int n = wave * 16 + r + 8 * group;
      lsh[n] = (n < NCOMBO) ? (pm + __logf(ps)) : -INFINITY;
    }
  }
  __syncthreads();

  if (wave == 0) {
    const float v0 = lsh[lane], v1 = lsh[lane + 32];
    const float v2 = lsh[lane + 64], v3 = lsh[lane + 96];
    float pm = fmaxf(fmaxf(v0, v1), fmaxf(v2, v3));
#pragma unroll
    for (int off = 1; off <= 16; off <<= 1)
      pm = fmaxf(pm, __shfl_xor(pm, off, 32));               // pm finite (real rows exist)
    float ps = __expf(v0 - pm) + __expf(v1 - pm) + __expf(v2 - pm) + __expf(v3 - pm);
#pragma unroll
    for (int off = 1; off <= 16; off <<= 1)
      ps += __shfl_xor(ps, off, 32);
    if (lane == 0) {
      const float gll = pm + __logf(ps) - __logf((float)NCOMBO);
      int nf = 0;
#pragma unroll
      for (int d = 0; d < ND_N; ++d) nf += (xs[d] != EMPTY_TOK) ? 1 : 0;
      const float fnf = (float)nf;
      const float rr  = rate[0];
      out[b] = fnf * rr - __expf(rr) - lgammaf(fnf + 1.f) + gll;
    }
  }
}

// -------------------------------------------------------------------
extern "C" void kernel_launch(void* const* d_in, const int* in_sizes, int n_in,
                              void* d_out, int out_size, void* d_ws, size_t ws_size,
                              hipStream_t stream) {
  const int*   x_g     = (const int*)d_in[0];    // [B,9] int32
  const int*   a_g     = (const int*)d_in[1];    // [B,9,9] int32
  const float* theta_g = (const float*)d_in[2];  // [40,30,5] f32
  const float* w_g     = (const float*)d_in[3];  // [40] f32
  const float* rate_g  = (const float*)d_in[4];  // [1] f32
  float*       out_g   = (float*)d_out;

  const int B = in_sizes[0] / ND_N;

  int*      ws_combos = (int*)((char*)d_ws + WS_COMBOS);
  float*    ws_lw     = (float*)((char*)d_ws + WS_LOGW);
  _Float16* ws_bsw    = (_Float16*)((char*)d_ws + WS_BSW);

  spn_prep_kernel<<<1, 256, 0, stream>>>(theta_g, w_g, ws_combos, ws_lw, ws_bsw);
  spn_main_kernel<<<B, 256, 0, stream>>>(x_g, a_g, rate_g, ws_combos, ws_lw, ws_bsw, out_g);
}
